// TransformerLayer_1365799600734
// MI455X (gfx1250) — matrix operand order repaired and checked by reading, not compiled, not run on hardware
//
#include <hip/hip_runtime.h>
#include <hip/hip_bf16.h>
#include <hip/hip_fp16.h>

// MI455X / gfx1250, wave32. All GEMMs via v_wmma_f32_16x16x32_f16.

typedef __attribute__((ext_vector_type(16))) _Float16 v16h;
typedef __attribute__((ext_vector_type(8)))  _Float16 v8h;
typedef __attribute__((ext_vector_type(8)))  float    v8f;

static constexpr int Bc = 64, Sc = 512, Dc = 512, Hc = 8, DKc = 64, DFFc = 2048;
static constexpr int Mrows = Bc * Sc;   // 32768

// ---------------------------------------------------------------------------
// WMMA fragment loaders (CDNA5 16-bit layouts, CDNA5 ISA §7.12.2)
// A (16x32, MxK): lane L holds row M=L%16; K = (e&7) + 8*(L>=16) + 16*(e>=8)
//   -> with row-major [m][k] LDS, two contiguous 8-half (16B) loads.
// B (32x16, KxN): lane L holds col N=L%16; K = (e&7) + 8*(L>=16) + 16*(e>=8)
//   -> with transposed [n][k] LDS, two contiguous 8-half (16B) loads.
// C/D (16x16 f32): vgpr r holds (M = r + 8*(L>=16), N = L%16).
// ---------------------------------------------------------------------------
__device__ __forceinline__ v16h frag_a(const _Float16* base, int lda, int mBase, int kBase) {
  const int lane = threadIdx.x & 31;
  const int half = lane >> 4;
  const int row  = mBase + (lane & 15);
  const _Float16* p = base + row * lda + kBase + half * 8;
  v8h lo = *(const v8h*)(p);
  v8h hi = *(const v8h*)(p + 16);
  v16h a;
#pragma unroll
  for (int e = 0; e < 8; ++e) { a[e] = lo[e]; a[e + 8] = hi[e]; }
  return a;
}

__device__ __forceinline__ v16h frag_b(const _Float16* base, int ldb, int nBase, int kBase) {
  const int lane = threadIdx.x & 31;
  const int half = lane >> 4;
  const int row  = nBase + (lane & 15);
  const _Float16* p = base + row * ldb + kBase + half * 8;
  v8h lo = *(const v8h*)(p);
  v8h hi = *(const v8h*)(p + 16);
  v16h b;
#pragma unroll
  for (int e = 0; e < 8; ++e) { b[e] = lo[e]; b[e + 8] = hi[e]; }
  return b;
}

__device__ __forceinline__ v8f wmma16(v16h a, v16h b, v8f c) {
  return __builtin_amdgcn_wmma_f32_16x16x32_f16(false, a, false, b, (short)0, c, false, false);
}

// ---------------------------------------------------------------------------
// GEMM: C[M,N] = A[M,K] @ W[K,N] + bias, optional ReLU.
// Block 256 thr = 8 waves (4 M x 2 N), tile 128x128, K-step 32.
// Double-buffered LDS: global loads for step k+1 issue before the 8 WMMAs of
// step k; LDS stores land after them -> HBM/L2 latency hidden by matrix math,
// one barrier per K-step.
// ---------------------------------------------------------------------------
__global__ __launch_bounds__(256) void gemm_bias_kernel(
    const float* __restrict__ A, const float* __restrict__ W,
    const float* __restrict__ bias, float* __restrict__ C,
    int M, int N, int K, int relu) {
  __shared__ __align__(16) _Float16 sA [2][128][40];   // [buf][m][k]
  __shared__ __align__(16) _Float16 sBt[2][128][40];   // [buf][n][k]

  const int tid  = threadIdx.x;
  const int wave = tid >> 5, lane = tid & 31, half = lane >> 4, col = lane & 15;
  const int wm = wave & 3, wn = wave >> 2;
  const int m0 = blockIdx.y * 128, n0 = blockIdx.x * 128;

  const int rA = tid >> 3;          // A stage: row 0..31 (+32*i)
  const int cA = (tid & 7) * 4;     // A stage: k-col
  const int kB = tid >> 5;          // B stage: k-row 0..7 (+8*i)
  const int nB = (tid & 31) * 4;    // B stage: n-col
  const float* Ap = A + (size_t)(m0 + rA) * K + cA;
  const float* Wp = W + (size_t)kB * N + n0 + nB;

  v8f acc[2][4] = {};

  // Prologue: stage K-tile 0 into buffer 0 (batched loads -> one wait).
  {
    float4 fa[4], fb[4];
#pragma unroll
    for (int i = 0; i < 4; ++i) fa[i] = *(const float4*)(Ap + (size_t)(i * 32) * K);
#pragma unroll
    for (int i = 0; i < 4; ++i) fb[i] = *(const float4*)(Wp + (size_t)(i * 8) * N);
#pragma unroll
    for (int i = 0; i < 4; ++i) {
      sA[0][rA + i * 32][cA + 0] = (_Float16)fa[i].x;
      sA[0][rA + i * 32][cA + 1] = (_Float16)fa[i].y;
      sA[0][rA + i * 32][cA + 2] = (_Float16)fa[i].z;
      sA[0][rA + i * 32][cA + 3] = (_Float16)fa[i].w;
      sBt[0][nB + 0][kB + i * 8] = (_Float16)fb[i].x;
      sBt[0][nB + 1][kB + i * 8] = (_Float16)fb[i].y;
      sBt[0][nB + 2][kB + i * 8] = (_Float16)fb[i].z;
      sBt[0][nB + 3][kB + i * 8] = (_Float16)fb[i].w;
    }
  }

  int p = 0;
  for (int k0 = 0; k0 < K; k0 += 32) {
    __syncthreads();
    const bool more = (k0 + 32) < K;
    float4 fa[4], fb[4];
    if (more) {
      // Issue next tile's global loads before this tile's WMMAs.
#pragma unroll
      for (int i = 0; i < 4; ++i)
        fa[i] = *(const float4*)(Ap + (size_t)(i * 32) * K + (k0 + 32));
#pragma unroll
      for (int i = 0; i < 4; ++i)
        fb[i] = *(const float4*)(Wp + (size_t)(k0 + 32 + i * 8) * N);
      if (k0 + 64 < K) {
        __builtin_prefetch(Ap + (size_t)0 * K + (k0 + 64), 0, 1);        // global_prefetch_b8
        __builtin_prefetch(Wp + (size_t)(k0 + 64) * N, 0, 1);
      }
    }

    // Compute from buffer p.
    const _Float16* aB = &sA [p][0][0];
    const _Float16* bB = &sBt[p][0][0];
    v16h a0 = frag_a(aB, 40, wm * 32,      0);
    v16h a1 = frag_a(aB, 40, wm * 32 + 16, 0);
    v16h b0 = frag_b(bB, 40, wn * 64,      0);
    v16h b1 = frag_b(bB, 40, wn * 64 + 16, 0);
    v16h b2 = frag_b(bB, 40, wn * 64 + 32, 0);
    v16h b3 = frag_b(bB, 40, wn * 64 + 48, 0);
    acc[0][0] = wmma16(a0, b0, acc[0][0]);
    acc[0][1] = wmma16(a0, b1, acc[0][1]);
    acc[0][2] = wmma16(a0, b2, acc[0][2]);
    acc[0][3] = wmma16(a0, b3, acc[0][3]);
    acc[1][0] = wmma16(a1, b0, acc[1][0]);
    acc[1][1] = wmma16(a1, b1, acc[1][1]);
    acc[1][2] = wmma16(a1, b2, acc[1][2]);
    acc[1][3] = wmma16(a1, b3, acc[1][3]);

    if (more) {
      const int q = p ^ 1;
#pragma unroll
      for (int i = 0; i < 4; ++i) {
        sA[q][rA + i * 32][cA + 0] = (_Float16)fa[i].x;
        sA[q][rA + i * 32][cA + 1] = (_Float16)fa[i].y;
        sA[q][rA + i * 32][cA + 2] = (_Float16)fa[i].z;
        sA[q][rA + i * 32][cA + 3] = (_Float16)fa[i].w;
        sBt[q][nB + 0][kB + i * 8] = (_Float16)fb[i].x;
        sBt[q][nB + 1][kB + i * 8] = (_Float16)fb[i].y;
        sBt[q][nB + 2][kB + i * 8] = (_Float16)fb[i].z;
        sBt[q][nB + 3][kB + i * 8] = (_Float16)fb[i].w;
      }
    }
    p ^= 1;
  }

#pragma unroll
  for (int tm = 0; tm < 2; ++tm) {
#pragma unroll
    for (int tn = 0; tn < 4; ++tn) {
      int n = n0 + wn * 64 + tn * 16 + col;
      float bvv = bias[n];
#pragma unroll
      for (int r = 0; r < 8; ++r) {
        int m = m0 + wm * 32 + tm * 16 + r + 8 * half;
        float vout = acc[tm][tn][r] + bvv;
        if (relu) vout = fmaxf(vout, 0.f);
        C[(size_t)m * N + n] = vout;
      }
    }
  }
}

// ---------------------------------------------------------------------------
// Flash-style causal attention with forget_rate. One block = (b, h, 64 q rows),
// 4 waves; each wave owns 16 query rows. Online softmax in f32, both matmuls
// via WMMA. Masked scores use -1e32 (matches the reference numerics), and
// query row 0 is zeroed when mask==0 (zero_pad).
// ---------------------------------------------------------------------------
__global__ __launch_bounds__(128) void attention_kernel(
    const float* __restrict__ q, const float* __restrict__ kk,
    const float* __restrict__ vv, const float* __restrict__ forget,
    float* __restrict__ attn, const int* __restrict__ maskp) {
  __shared__ __align__(16) _Float16 sQ [64][72];   // [q_row][dk]   (A for QK^T)
  __shared__ __align__(16) _Float16 sK [64][72];   // [key][dk]     (Bt for QK^T)
  __shared__ __align__(16) _Float16 sVt[64][72];   // [dk][key]     (Bt for PV)
  __shared__ __align__(16) _Float16 sP [64][72];   // [q_row][key]  (A for PV)

  const int tid  = threadIdx.x;
  const int wave = tid >> 5, lane = tid & 31, half = lane >> 4, col = lane & 15;
  const int q0 = blockIdx.x * 64;
  const int bh = blockIdx.y;
  const int b  = bh >> 3, h = bh & 7;
  const int msk = *maskp;
  const size_t rowBase = (size_t)b * Sc;
  const int rS  = tid >> 4;          // staging row 0..7 (+8*i)
  const int cS  = (tid & 15) * 4;    // staging col

  // Stage Q tile 64x64 (f32 -> f16), batched loads then stores.
#pragma unroll
  for (int g = 0; g < 2; ++g) {
    float4 f[4];
#pragma unroll
    for (int i = 0; i < 4; ++i)
      f[i] = *(const float4*)(q + (rowBase + q0 + rS + (g * 4 + i) * 8) * Dc + h * DKc + cS);
#pragma unroll
    for (int i = 0; i < 4; ++i) {
      int r = rS + (g * 4 + i) * 8;
      sQ[r][cS + 0] = (_Float16)f[i].x; sQ[r][cS + 1] = (_Float16)f[i].y;
      sQ[r][cS + 2] = (_Float16)f[i].z; sQ[r][cS + 3] = (_Float16)f[i].w;
    }
  }

  const int mBase = wave * 16;
  float m_i[8], l_i[8], fr[8];
  v8f acc[4] = {};
#pragma unroll
  for (int r = 0; r < 8; ++r) {
    m_i[r] = -__builtin_inff();
    l_i[r] = 0.f;
    // fold 1/sqrt(DK)=0.125 into per-row forget rate
    fr[r] = forget[rowBase + q0 + mBase + r + 8 * half] * 0.125f;
  }

  int jEnd = q0 + 64 + msk;            // causal: only key tiles with j < i+mask
  if (jEnd > Sc) jEnd = Sc;

  for (int j0 = 0; j0 < jEnd; j0 += 64) {
    __syncthreads();
    // Stage K tile [key][dk] and V tile transposed [dk][key]; batched loads.
#pragma unroll
    for (int g = 0; g < 2; ++g) {
      float4 fk[4], fv[4];
#pragma unroll
      for (int i = 0; i < 4; ++i) {
        size_t gr = (rowBase + j0 + rS + (g * 4 + i) * 8) * Dc + h * DKc + cS;
        fk[i] = *(const float4*)(kk + gr);
        fv[i] = *(const float4*)(vv + gr);
      }
#pragma unroll
      for (int i = 0; i < 4; ++i) {
        int r = rS + (g * 4 + i) * 8;
        sK[r][cS + 0] = (_Float16)fk[i].x; sK[r][cS + 1] = (_Float16)fk[i].y;
        sK[r][cS + 2] = (_Float16)fk[i].z; sK[r][cS + 3] = (_Float16)fk[i].w;
        sVt[cS + 0][r] = (_Float16)fv[i].x; sVt[cS + 1][r] = (_Float16)fv[i].y;
        sVt[cS + 2][r] = (_Float16)fv[i].z; sVt[cS + 3][r] = (_Float16)fv[i].w;
      }
    }
    __syncthreads();

    // S = Q @ K^T  (16 q rows x 64 keys per wave)
    v8f sc[4] = {};
#pragma unroll
    for (int ks = 0; ks < 2; ++ks) {
      v16h aq = frag_a(&sQ[0][0], 72, mBase, ks * 32);
#pragma unroll
      for (int nt = 0; nt < 4; ++nt) {
        v16h bk = frag_b(&sK[0][0], 72, nt * 16, ks * 32);
        sc[nt] = wmma16(aq, bk, sc[nt]);
      }
    }

    // Online softmax per row (8 rows per lane; row data across 16 lanes).
#pragma unroll
    for (int r = 0; r < 8; ++r) {
      const int iLoc  = mBase + r + 8 * half;
      const int iGlob = q0 + iLoc;
      float vals[4];
      float vmax = -__builtin_inff();
#pragma unroll
      for (int nt = 0; nt < 4; ++nt) {
        int jGlob = j0 + nt * 16 + col;
        float sv = sc[nt][r] * fr[r];
        sv = (jGlob < iGlob + msk) ? sv : -1e32f;
        vals[nt] = sv;
        vmax = fmaxf(vmax, sv);
      }
#pragma unroll
      for (int off = 1; off < 16; off <<= 1)
        vmax = fmaxf(vmax, __shfl_xor(vmax, off, 32));
      float mNew = fmaxf(m_i[r], vmax);
      float corr = __expf(m_i[r] - mNew);      // first iter: exp(-inf)=0
      float rsum = 0.f;
#pragma unroll
      for (int nt = 0; nt < 4; ++nt) {
        float pv = __expf(vals[nt] - mNew);
        rsum += pv;
        sP[iLoc][nt * 16 + col] = (_Float16)pv;
      }
#pragma unroll
      for (int off = 1; off < 16; off <<= 1)
        rsum += __shfl_xor(rsum, off, 32);
      l_i[r] = l_i[r] * corr + rsum;
      m_i[r] = mNew;
#pragma unroll
      for (int nt = 0; nt < 4; ++nt) acc[nt][r] = acc[nt][r] * corr;
    }

    __syncthreads();
    // O += P @ V  (16 q rows x 64 dk per wave)
#pragma unroll
    for (int ks = 0; ks < 2; ++ks) {
      v16h ap = frag_a(&sP[0][0], 72, mBase, ks * 32);
#pragma unroll
      for (int nt = 0; nt < 4; ++nt) {
        v16h bv = frag_b(&sVt[0][0], 72, nt * 16, ks * 32);
        acc[nt] = wmma16(ap, bv, acc[nt]);
      }
    }
  }

  // Epilogue: divide by l, zero_pad row 0, write concat layout [b*S+i][h*64+d].
#pragma unroll
  for (int r = 0; r < 8; ++r) {
    int iGlob = q0 + mBase + r + 8 * half;
    float scale = (l_i[r] > 0.f) ? (1.f / l_i[r]) : 0.f;
    if (msk == 0 && iGlob == 0) scale = 0.f;
#pragma unroll
    for (int nt = 0; nt < 4; ++nt) {
      attn[(rowBase + iGlob) * Dc + h * DKc + nt * 16 + col] = acc[nt][r] * scale;
    }
  }
}

// ---------------------------------------------------------------------------
// out = LayerNorm(x + y) * g + beta  (D=512, wave-per-row, float4 I/O,
// shuffle reductions; biased variance matches jnp.var)
// ---------------------------------------------------------------------------
__global__ __launch_bounds__(256) void add_ln_kernel(
    const float* __restrict__ x, const float* __restrict__ y,
    const float* __restrict__ g, const float* __restrict__ beta,
    float* __restrict__ out) {
  const int wave = threadIdx.x >> 5, lane = threadIdx.x & 31;
  const size_t row = (size_t)blockIdx.x * 8 + wave;
  const float4* x4 = (const float4*)(x + row * Dc);
  const float4* y4 = (const float4*)(y + row * Dc);

  float4 vbuf[4];
  float s1 = 0.f, s2 = 0.f;
#pragma unroll
  for (int j = 0; j < 4; ++j) {
    float4 a = x4[lane * 4 + j];
    float4 bb = y4[lane * 4 + j];
    a.x += bb.x; a.y += bb.y; a.z += bb.z; a.w += bb.w;
    vbuf[j] = a;
    s1 += a.x + a.y + a.z + a.w;
    s2 += a.x * a.x + a.y * a.y + a.z * a.z + a.w * a.w;
  }
#pragma unroll
  for (int off = 1; off < 32; off <<= 1) {
    s1 += __shfl_xor(s1, off, 32);
    s2 += __shfl_xor(s2, off, 32);
  }
  float mu  = s1 * (1.0f / Dc);
  float var = s2 * (1.0f / Dc) - mu * mu;
  float inv = rsqrtf(var + 1e-5f);

  const float4* g4 = (const float4*)g;
  const float4* b4 = (const float4*)beta;
  float4* o4 = (float4*)(out + row * Dc);
#pragma unroll
  for (int j = 0; j < 4; ++j) {
    float4 gg = g4[lane * 4 + j];
    float4 be = b4[lane * 4 + j];
    float4 a  = vbuf[j];
    float4 o;
    o.x = (a.x - mu) * inv * gg.x + be.x;
    o.y = (a.y - mu) * inv * gg.y + be.y;
    o.z = (a.z - mu) * inv * gg.z + be.z;
    o.w = (a.w - mu) * inv * gg.w + be.w;
    o4[lane * 4 + j] = o;
  }
}

// ---------------------------------------------------------------------------
extern "C" void kernel_launch(void* const* d_in, const int* in_sizes, int n_in,
                              void* d_out, int out_size, void* d_ws, size_t ws_size,
                              hipStream_t stream) {
  (void)in_sizes; (void)n_in; (void)out_size; (void)ws_size;

  const float* query  = (const float*)d_in[0];
  const float* key_   = (const float*)d_in[1];
  const float* vals   = (const float*)d_in[2];
  const float* forget = (const float*)d_in[3];
  const float* Wq = (const float*)d_in[4];  const float* bq  = (const float*)d_in[5];
  const float* Wk = (const float*)d_in[6];  const float* bk  = (const float*)d_in[7];
  const float* Wv = (const float*)d_in[8];  const float* bv  = (const float*)d_in[9];
  const float* Wo = (const float*)d_in[10]; const float* bo  = (const float*)d_in[11];
  const float* g1 = (const float*)d_in[12]; const float* be1 = (const float*)d_in[13];
  const float* W1 = (const float*)d_in[14]; const float* bf1 = (const float*)d_in[15];
  const float* W2 = (const float*)d_in[16]; const float* bf2 = (const float*)d_in[17];
  const float* g2 = (const float*)d_in[18]; const float* be2 = (const float*)d_in[19];
  const int*  maskp = (const int*)d_in[20];

  float* out = (float*)d_out;
  float* ws  = (float*)d_ws;

  // Workspace layout (floats): q,k,v,attn = Mrows*D each; ffh = Mrows*DFF.
  // proj/x1/ff reuse the q/k/v regions once they are dead. Total 512 MB.
  const size_t SZ = (size_t)Mrows * Dc;
  float* qb   = ws;
  float* kb   = ws + SZ;
  float* vb   = ws + 2 * SZ;
  float* attn = ws + 3 * SZ;
  float* ffh  = ws + 4 * SZ;       // Mrows x DFF (4*SZ floats)
  float* proj = qb;                // reuse (q dead after attention)
  float* x1   = kb;                // reuse (k dead after attention)
  float* ffo  = vb;                // reuse (v dead after attention)

  dim3 gProj(Dc / 128, Mrows / 128);       // (4, 256)
  dim3 gFF1(DFFc / 128, Mrows / 128);      // (16, 256)

  gemm_bias_kernel<<<gProj, 256, 0, stream>>>(query, Wq, bq, qb, Mrows, Dc, Dc, 0);
  gemm_bias_kernel<<<gProj, 256, 0, stream>>>(key_,  Wk, bk, kb, Mrows, Dc, Dc, 0);
  gemm_bias_kernel<<<gProj, 256, 0, stream>>>(vals,  Wv, bv, vb, Mrows, Dc, Dc, 0);

  attention_kernel<<<dim3(Sc / 64, Bc * Hc), 128, 0, stream>>>(qb, kb, vb, forget, attn, maskp);

  gemm_bias_kernel<<<gProj, 256, 0, stream>>>(attn, Wo, bo, proj, Mrows, Dc, Dc, 0);
  add_ln_kernel<<<Mrows / 8, 256, 0, stream>>>(query, proj, g1, be1, x1);

  gemm_bias_kernel<<<gFF1, 256, 0, stream>>>(x1,  W1, bf1, ffh, Mrows, DFFc, Dc, 1);
  gemm_bias_kernel<<<gProj, 256, 0, stream>>>(ffh, W2, bf2, ffo, Mrows, Dc, DFFc, 0);
  add_ln_kernel<<<Mrows / 8, 256, 0, stream>>>(x1, ffo, g2, be2, out);
}
